// Backbone_1735166788084
// MI455X (gfx1250) — matrix operand order repaired and checked by reading, not compiled, not run on hardware
//
#include <hip/hip_runtime.h>

// ============================ CDNA5 / gfx1250 notes =========================
// - wave32 only; 256-thread blocks = 8 waves.
// - Matrix ops: v_wmma_f32_16x16x32_f16 (f16 in, f32 accumulate).
// - Roofline: the 3 trailing MLP GEMMs are ~33 GFLOP (>95% of total FLOPs);
//   total HBM traffic ~150MB -> ~7us @ 23.3TB/s => GEMMs are compute bound and
//   ride the WMMA path. A-tiles are staged with CDNA5 async global->LDS DMA
//   (global_load_async_to_lds_b128 / s_wait_asynccnt), B-tiles are staged in a
//   pair-interleaved layout so each WMMA B-fragment VGPR is a single
//   ds_load_b32. LDS tiles are double buffered.
// ===========================================================================

#define BB 4
#define NN 2048
#define EPSV 1e-6f
#define BNEPS 1e-5f
#define NEGS 0.2f

typedef __attribute__((ext_vector_type(16))) _Float16 v16h;
typedef __attribute__((ext_vector_type(8)))  float    v8f;

// ----------------------------- utility kernels -----------------------------
__global__ void zero_f32_k(float* p, int n) {
  int i = blockIdx.x * blockDim.x + threadIdx.x;
  if (i < n) p[i] = 0.f;
}
__global__ void zero_f16_k(_Float16* p, int n) {
  int i = blockIdx.x * blockDim.x + threadIdx.x;
  if (i < n) p[i] = (_Float16)0.f;
}

// ------------------------------- KNN (top-K) -------------------------------
template <int K>
__global__ void knn_kernel(const float* __restrict__ coord, int Ncand, int cstride,
                           int qstride, int Nq, int* __restrict__ idx) {
  int tid = blockIdx.x * blockDim.x + threadIdx.x;
  if (tid >= BB * Nq) return;
  int b = tid / Nq, q = tid - b * Nq;
  const float* cb = coord + (size_t)b * 3 * NN;
  float qx = cb[q * qstride], qy = cb[NN + q * qstride], qz = cb[2 * NN + q * qstride];
  float qq = qx * qx + qy * qy + qz * qz;
  float bd[K];
  int   bi[K];
#pragma unroll
  for (int k = 0; k < K; ++k) { bd[k] = -3.4e38f; bi[k] = 0; }
  for (int j = 0; j < Ncand; ++j) {
    float sx = cb[j * cstride], sy = cb[NN + j * cstride], sz = cb[2 * NN + j * cstride];
    float pd = 2.f * (qx * sx + qy * sy + qz * sz) - qq - (sx * sx + sy * sy + sz * sz);
    if (pd > bd[K - 1]) {
      int k = K - 1;
      while (k > 0 && bd[k - 1] < pd) { bd[k] = bd[k - 1]; bi[k] = bi[k - 1]; --k; }
      bd[k] = pd; bi[k] = j;
    }
  }
#pragma unroll
  for (int k = 0; k < K; ++k) idx[(size_t)(b * Nq + q) * K + k] = bi[k];
}

// --------------------------- nearest index (argmin) -------------------------
__global__ void nearest_kernel(const float* __restrict__ coord, int Ns, int sstride,
                               int* __restrict__ out) {
  int tid = blockIdx.x * blockDim.x + threadIdx.x;
  if (tid >= BB * NN) return;
  int b = tid >> 11, i = tid & (NN - 1);
  const float* cb = coord + (size_t)b * 3 * NN;
  float tx = cb[i], ty = cb[NN + i], tz = cb[2 * NN + i];
  float tt = tx * tx + ty * ty + tz * tz;
  float best = 3.4e38f; int bj = 0;
  for (int j = 0; j < Ns; ++j) {
    float sx = cb[j * sstride], sy = cb[NN + j * sstride], sz = cb[2 * NN + j * sstride];
    float d = tt - 2.f * (tx * sx + ty * sy + tz * sz) + sx * sx + sy * sy + sz * sz;
    if (d < best) { best = d; bj = j; }
  }
  out[tid] = bj;
}

// -------------------- surface feature block (x0) ---------------------------
__device__ __forceinline__ void surf_pd(const float* cb, int i, int j,
                                        const float* w3, float pv[3]) {
  float cx = cb[i], cy = cb[NN + i], cz = cb[2 * NN + i];
  float nx = cb[j], ny = cb[NN + j], nz = cb[2 * NN + j];
  float f0x = nx - cx, f0y = ny - cy, f0z = nz - cz;
  float f2x = ny * cz - nz * cy, f2y = nz * cx - nx * cz, f2z = nx * cy - ny * cx;
  pv[0] = w3[0] * f0x + w3[1] * cx + w3[2] * f2x;
  pv[1] = w3[0] * f0y + w3[1] * cy + w3[2] * f2y;
  pv[2] = w3[0] * f0z + w3[1] * cz + w3[2] * f2z;
}

__global__ void surf_stats_k(const float* __restrict__ coord, const int* __restrict__ idx20,
                             const float* __restrict__ wf0,
                             float* __restrict__ ssum, float* __restrict__ ssq) {
  __shared__ float rs[256], rq[256];
  int o = blockIdx.x, b = blockIdx.y;
  const float* cb = coord + (size_t)b * 3 * NN;
  const float* w3 = wf0 + o * 3;
  float ls = 0.f, lq = 0.f;
  for (int t = threadIdx.x; t < NN * 20; t += blockDim.x) {
    int i = t / 20, kk = t - i * 20;
    int j = idx20[(size_t)(b * NN + i) * 20 + kk];
    float pv[3];
    surf_pd(cb, i, j, w3, pv);
    float nn = sqrtf(pv[0] * pv[0] + pv[1] * pv[1] + pv[2] * pv[2]) + EPSV;
    ls += nn; lq += nn * nn;
  }
  rs[threadIdx.x] = ls; rq[threadIdx.x] = lq; __syncthreads();
  for (int st = 128; st > 0; st >>= 1) {
    if (threadIdx.x < st) { rs[threadIdx.x] += rs[threadIdx.x + st]; rq[threadIdx.x] += rq[threadIdx.x + st]; }
    __syncthreads();
  }
  if (threadIdx.x == 0) { atomicAdd(&ssum[o], rs[0]); atomicAdd(&ssq[o], rq[0]); }
}

__global__ void surf_apply_k(const float* __restrict__ coord, const int* __restrict__ idx20,
                             const float* __restrict__ wf0, const float* __restrict__ wd0,
                             const float* __restrict__ g0, const float* __restrict__ b0,
                             const float* __restrict__ m, const float* __restrict__ inv,
                             float* __restrict__ x0) {
  int o = blockIdx.x, b = blockIdx.y;
  const float* cb = coord + (size_t)b * 3 * NN;
  const float* wfo = wf0 + o * 3;
  const float* wdo = wd0 + o * 3;
  float gg = g0[o], bb = b0[o], mm = m[o], iv = inv[o];
  for (int n = threadIdx.x; n < NN; n += blockDim.x) {
    float acc[3] = {0.f, 0.f, 0.f};
    for (int kk = 0; kk < 20; ++kk) {
      int j = idx20[(size_t)(b * NN + n) * 20 + kk];
      float pv[3], dv[3];
      surf_pd(cb, n, j, wfo, pv);
      surf_pd(cb, n, j, wdo, dv);
      float nn = sqrtf(pv[0] * pv[0] + pv[1] * pv[1] + pv[2] * pv[2]) + EPSV;
      float nb = gg * (nn - mm) * iv + bb;
      float q[3] = {pv[0] / nn * nb, pv[1] / nn * nb, pv[2] / nn * nb};
      float dot = q[0] * dv[0] + q[1] * dv[1] + q[2] * dv[2];
      float dsq = dv[0] * dv[0] + dv[1] * dv[1] + dv[2] * dv[2];
      float s = (dot >= 0.f) ? 0.f : dot / (dsq + EPSV);
#pragma unroll
      for (int v = 0; v < 3; ++v) {
        float w = q[v] - s * dv[v];
        acc[v] += NEGS * q[v] + (1.f - NEGS) * w;
      }
    }
#pragma unroll
    for (int v = 0; v < 3; ++v)
      x0[((size_t)(b * 42 + o) * 3 + v) * NN + n] = acc[v] * (1.f / 20.f);
  }
}

// ----------------------- generic VN block (p,d + stats) ---------------------
__global__ void vn_lin_pair_k(const float* __restrict__ x, const float* __restrict__ wf,
                              const float* __restrict__ wd, float* __restrict__ p,
                              float* __restrict__ dv, float* __restrict__ ssum,
                              float* __restrict__ ssq, int Cin, int Cout, int Np) {
  __shared__ float rs[256], rq[256];
  int o = blockIdx.x, b = blockIdx.y;
  float ls = 0.f, lq = 0.f;
  for (int n = threadIdx.x; n < Np; n += blockDim.x) {
    float pv[3] = {0.f, 0.f, 0.f}, dd[3] = {0.f, 0.f, 0.f};
    for (int c = 0; c < Cin; ++c) {
      float a = wf[o * Cin + c], e = wd[o * Cin + c];
#pragma unroll
      for (int v = 0; v < 3; ++v) {
        float xv = x[((size_t)(b * Cin + c) * 3 + v) * Np + n];
        pv[v] += a * xv; dd[v] += e * xv;
      }
    }
#pragma unroll
    for (int v = 0; v < 3; ++v) {
      p[((size_t)(b * Cout + o) * 3 + v) * Np + n] = pv[v];
      dv[((size_t)(b * Cout + o) * 3 + v) * Np + n] = dd[v];
    }
    float nn = sqrtf(pv[0] * pv[0] + pv[1] * pv[1] + pv[2] * pv[2]) + EPSV;
    ls += nn; lq += nn * nn;
  }
  rs[threadIdx.x] = ls; rq[threadIdx.x] = lq; __syncthreads();
  for (int st = 128; st > 0; st >>= 1) {
    if (threadIdx.x < st) { rs[threadIdx.x] += rs[threadIdx.x + st]; rq[threadIdx.x] += rq[threadIdx.x + st]; }
    __syncthreads();
  }
  if (threadIdx.x == 0) { atomicAdd(&ssum[o], rs[0]); atomicAdd(&ssq[o], rq[0]); }
}

__global__ void stats_fin_k(const float* ssum, const float* ssq, float* m, float* inv,
                            int C, float cnt) {
  int o = blockIdx.x * blockDim.x + threadIdx.x;
  if (o >= C) return;
  float mm = ssum[o] / cnt;
  float var = ssq[o] / cnt - mm * mm;
  m[o] = mm;
  inv[o] = rsqrtf(var + BNEPS);
}

__global__ void vn_bn_leaky_k(const float* __restrict__ p, const float* __restrict__ dv,
                              const float* __restrict__ g, const float* __restrict__ bb,
                              const float* __restrict__ m, const float* __restrict__ inv,
                              float* __restrict__ out, int Cout, int Np) {
  int bo = blockIdx.x;
  int b = bo / Cout, o = bo - b * Cout;
  float gg = g[o], be = bb[o], mm = m[o], iv = inv[o];
  for (int n = threadIdx.x; n < Np; n += blockDim.x) {
    float pv[3], dd[3];
#pragma unroll
    for (int v = 0; v < 3; ++v) {
      pv[v] = p[((size_t)(b * Cout + o) * 3 + v) * Np + n];
      dd[v] = dv[((size_t)(b * Cout + o) * 3 + v) * Np + n];
    }
    float nn = sqrtf(pv[0] * pv[0] + pv[1] * pv[1] + pv[2] * pv[2]) + EPSV;
    float nb = gg * (nn - mm) * iv + be;
    float q[3] = {pv[0] / nn * nb, pv[1] / nn * nb, pv[2] / nn * nb};
    float dot = q[0] * dd[0] + q[1] * dd[1] + q[2] * dd[2];
    float dsq = dd[0] * dd[0] + dd[1] * dd[1] + dd[2] * dd[2];
    float s = (dot >= 0.f) ? 0.f : dot / (dsq + EPSV);
#pragma unroll
    for (int v = 0; v < 3; ++v) {
      float w = q[v] - s * dd[v];
      out[((size_t)(b * Cout + o) * 3 + v) * Np + n] = NEGS * q[v] + (1.f - NEGS) * w;
    }
  }
}

// ----------------------- neighbor gather + VN max-pool ----------------------
__global__ void pool_down_k(const float* __restrict__ x, const int* __restrict__ idx,
                            const float* __restrict__ wp, float* __restrict__ out,
                            int C, int Ns, int Nd) {
  __shared__ float xg[4 * 84 * 3];
  int blk = blockIdx.x;
  int b = blk / Nd, nd = blk - b * Nd;
  int t = threadIdx.x;
  if (t < C) {
    for (int kk = 0; kk < 4; ++kk) {
      int j = idx[(size_t)(b * Nd + nd) * 4 + kk];
#pragma unroll
      for (int v = 0; v < 3; ++v)
        xg[(kk * C + t) * 3 + v] = x[((size_t)(b * C + t) * 3 + v) * Ns + j];
    }
  }
  __syncthreads();
  if (t < C) {
    float bestDot = -3.4e38f; int bestK = 0;
    for (int kk = 0; kk < 4; ++kk) {
      float p0 = 0.f, p1 = 0.f, p2 = 0.f;
      for (int c = 0; c < C; ++c) {
        float w = wp[t * C + c];
        p0 += w * xg[(kk * C + c) * 3 + 0];
        p1 += w * xg[(kk * C + c) * 3 + 1];
        p2 += w * xg[(kk * C + c) * 3 + 2];
      }
      float dot = xg[(kk * C + t) * 3 + 0] * p0 + xg[(kk * C + t) * 3 + 1] * p1 +
                  xg[(kk * C + t) * 3 + 2] * p2;
      if (dot > bestDot) { bestDot = dot; bestK = kk; }
    }
#pragma unroll
    for (int v = 0; v < 3; ++v)
      out[((size_t)(b * C + t) * 3 + v) * Nd + nd] = xg[(bestK * C + t) * 3 + v];
  }
}

// ------------------------- eqv assembly / mean_feat -------------------------
__global__ void eqv_assemble_k(const float* __restrict__ x0, const float* __restrict__ x1,
                               const float* __restrict__ x2, const float* __restrict__ x3,
                               const float* __restrict__ x4, const int* __restrict__ i1,
                               const int* __restrict__ i2, float* __restrict__ eqv) {
  int idx = blockIdx.x * blockDim.x + threadIdx.x;
  if (idx >= BB * 420 * NN) return;
  int n = idx & (NN - 1);
  int rest = idx >> 11;
  int ch = rest % 420, b = rest / 420;
#pragma unroll
  for (int v = 0; v < 3; ++v) {
    float val;
    if (ch < 42) val = x0[((size_t)(b * 42 + ch) * 3 + v) * NN + n];
    else if (ch < 84) val = x1[((size_t)(b * 42 + (ch - 42)) * 3 + v) * NN + n];
    else if (ch < 168) { int j = i1[b * NN + n]; val = x2[((size_t)(b * 84 + (ch - 84)) * 3 + v) * 512 + j]; }
    else if (ch < 252) { int j = i1[b * NN + n]; val = x3[((size_t)(b * 84 + (ch - 168)) * 3 + v) * 512 + j]; }
    else { int j = i2[b * NN + n]; val = x4[((size_t)(b * 168 + (ch - 252)) * 3 + v) * 128 + j]; }
    eqv[((size_t)(b * 420 + ch) * 3 + v) * NN + n] = val;
  }
}

__global__ void mean_feat_k(const float* __restrict__ eqv, float* __restrict__ mf) {
  __shared__ float rs[128];
  int cid = blockIdx.x;  // b*1260 + c*3 + v
  const float* src = eqv + (size_t)cid * NN;
  float s = 0.f;
  for (int n = threadIdx.x; n < NN; n += 128) s += src[n];
  rs[threadIdx.x] = s; __syncthreads();
  for (int st = 64; st > 0; st >>= 1) {
    if (threadIdx.x < st) rs[threadIdx.x] += rs[threadIdx.x + st];
    __syncthreads();
  }
  if (threadIdx.x == 0) mf[cid] = rs[0] * (1.f / (float)NN);
}

// ----------------------------- tiny z-pipeline ------------------------------
__global__ void vn_block_small_k(const float* __restrict__ xin, const float* __restrict__ wf,
                                 const float* __restrict__ wd, float* __restrict__ out,
                                 int Cin, int Cout) {
  int idx = blockIdx.x * blockDim.x + threadIdx.x;
  if (idx >= BB * Cout) return;
  int b = idx / Cout, o = idx - b * Cout;
  float pv[3] = {0.f, 0.f, 0.f}, dd[3] = {0.f, 0.f, 0.f};
  for (int c = 0; c < Cin; ++c) {
    float a = wf[o * Cin + c], e = wd[o * Cin + c];
#pragma unroll
    for (int v = 0; v < 3; ++v) {
      float xv = xin[(size_t)(b * Cin + c) * 3 + v];
      pv[v] += a * xv; dd[v] += e * xv;
    }
  }
  float dot = pv[0] * dd[0] + pv[1] * dd[1] + pv[2] * dd[2];
  float dsq = dd[0] * dd[0] + dd[1] * dd[1] + dd[2] * dd[2];
  float s = (dot >= 0.f) ? 0.f : dot / (dsq + EPSV);
#pragma unroll
  for (int v = 0; v < 3; ++v) {
    float w = pv[v] - s * dd[v];
    out[(size_t)(b * Cout + o) * 3 + v] = NEGS * pv[v] + (1.f - NEGS) * w;
  }
}

__global__ void inv_gl_k(const float* __restrict__ mf, const float* __restrict__ z2,
                         const float* __restrict__ w3, float* __restrict__ invgl) {
  int idx = blockIdx.x * blockDim.x + threadIdx.x;
  if (idx >= BB * 420) return;
  int b = idx / 420, i = idx - b * 420;
  float z3[3][2];
#pragma unroll
  for (int v = 0; v < 3; ++v)
#pragma unroll
    for (int kk = 0; kk < 2; ++kk) {
      float s = 0.f;
      for (int c = 0; c < 84; ++c) s += z2[(size_t)(b * 84 + c) * 3 + v] * w3[kk * 84 + c];
      z3[v][kk] = s;
    }
#pragma unroll
  for (int kk = 0; kk < 2; ++kk) {
    float s = 0.f;
#pragma unroll
    for (int v = 0; v < 3; ++v) s += mf[(size_t)b * 1260 + i * 3 + v] * z3[v][kk];
    invgl[(size_t)b * 840 + i * 2 + kk] = s;
  }
}

// ----------------------- invariant-head input (f16) -------------------------
__global__ void build_inv_input_k(const float* __restrict__ norm, const int* __restrict__ cat,
                                  const float* __restrict__ mf, const float* __restrict__ invgl,
                                  const float* __restrict__ coord, _Float16* __restrict__ X) {
  const int M = BB * NN;
  int idx = blockIdx.x * blockDim.x + threadIdx.x;
  if (idx >= 1267 * M) return;
  int r = idx / M, m = idx - r * M;
  int b = m >> 11, n = m & (NN - 1);
  float v;
  if (r == 0) v = norm[b * NN + n];
  else if (r < 841) v = invgl[(size_t)b * 840 + (r - 1)];
  else if (r < 1261) {
    int c = r - 841;
    float s = 0.f;
#pragma unroll
    for (int v3 = 0; v3 < 3; ++v3)
      s += mf[(size_t)b * 1260 + c * 3 + v3] * coord[((size_t)b * 3 + v3) * NN + n];
    v = s;
  } else {
    v = (cat[b] == (r - 1261)) ? 1.f : 0.f;
  }
  X[(size_t)r * M + m] = (_Float16)v;
}

__global__ void pad_w_k(const float* __restrict__ w, _Float16* __restrict__ wh,
                        int O, int C, int Opad, int Cpad) {
  int i = blockIdx.x * blockDim.x + threadIdx.x;
  if (i >= Opad * Cpad) return;
  int r = i / Cpad, c = i - r * Cpad;
  float v = (r < O && c < C) ? w[(size_t)r * C + c] : 0.f;
  wh[i] = (_Float16)v;
}

// ------------------------- WMMA GEMM (f16 -> f32) ---------------------------
// C[Opad x M] = A[Opad x K] * B[K x M]; 64x64 tile / block; 8 waves (4x2),
// each wave = 16x32 via two v_wmma_f32_16x16x32_f16 accumulators.
//
// A tile (64x32 f16) staged by async global->LDS DMA (one b128 per thread,
// tracked by ASYNCcnt). LDS row stride = 40 halves (80B = 5*16B) so every
// async 16B chunk is 16B-aligned in LDS.
//
// B tile staged pair-interleaved: one 32-bit LDS word = {K=2k+1, K=2k} halves
// at a given column, so each B-fragment VGPR is a single ds_load_b32.
// Tiles are double buffered; staging of tile k+1 overlaps WMMA on tile k.
#define LDA 40    // halves per A row (16B-aligned rows)
#define LDBW 68   // uint words per B row-pair

__global__ __launch_bounds__(256) void gemm_f16_wmma_k(
    const _Float16* __restrict__ A, const _Float16* __restrict__ Bm,
    float* __restrict__ C, int K, int M) {
  __shared__ _Float16 As[2][64 * LDA];
  __shared__ unsigned int Bsp[2][16 * LDBW];
  const int t = threadIdx.x;
  const int tileM = blockIdx.x * 64;
  const int tileO = blockIdx.y * 64;
  const int lane = t & 31;
  const int wv = t >> 5;
  const int wr = wv & 3;   // wave row block (16 rows each)
  const int wc = wv >> 2;  // wave col block (32 cols each)
  const int l16 = lane & 15;
  const int hh = lane >> 4;

  // staging coordinates
  const int ar = t >> 2;          // A row 0..63
  const int ac8 = (t & 3) << 3;   // A col group (8 halves = 16B)
  const int br2 = t >> 4;         // B row-pair 0..15
  const int bc4 = (t & 15) << 2;  // B col group (4 cols)

  const unsigned ldsA0 = (unsigned)(size_t)(void*)&As[0][0];
  const unsigned ldsA1 = (unsigned)(size_t)(void*)&As[1][0];

  auto stageA = [&](int k0, int buf) {
    unsigned dst = (buf ? ldsA1 : ldsA0) + (unsigned)((ar * LDA + ac8) * 2);
    unsigned long long src =
        (unsigned long long)(const void*)(A + (size_t)(tileO + ar) * K + k0 + ac8);
    asm volatile("global_load_async_to_lds_b128 %0, %1, off"
                 :: "v"(dst), "v"(src) : "memory");
  };
  auto stageB = [&](int k0, int buf) {
    const _Float16* re = Bm + (size_t)(k0 + 2 * br2) * M + tileM + bc4;
    const uint2 e = *(const uint2*)re;        // even K row, 4 cols
    const uint2 o = *(const uint2*)(re + M);  // odd  K row, 4 cols
    unsigned int* dst = &Bsp[buf][br2 * LDBW + bc4];
    dst[0] = (e.x & 0xFFFFu) | (o.x << 16);
    dst[1] = (e.x >> 16) | (o.x & 0xFFFF0000u);
    dst[2] = (e.y & 0xFFFFu) | (o.y << 16);
    dst[3] = (e.y >> 16) | (o.y & 0xFFFF0000u);
  };

  v8f acc0 = {};
  v8f acc1 = {};
  union VH { v16h v; _Float16 h[16]; unsigned int u[8]; };

  stageA(0, 0);
  stageB(0, 0);
  asm volatile("s_wait_asynccnt 0x0" ::: "memory");
  __syncthreads();

  int cur = 0;
  for (int k0 = 0; k0 < K; k0 += 32) {
    const int nxt = cur ^ 1;
    if (k0 + 32 < K) {  // stage next tile while computing this one
      stageA(k0 + 32, nxt);
      stageB(k0 + 32, nxt);
    }

    // A fragment (16x32 f16): lane row = l16, K-pairs in contiguous order k = 16*hh + s (same as B).
    VH af;
#pragma unroll
    for (int v = 0; v < 8; ++v) {
      int k = 2 * v + 16 * hh;
      af.u[v] = *(const unsigned int*)&As[cur][(wr * 16 + l16) * LDA + k];
    }
    // B fragments: one ds_load_b32 per VGPR (pair-interleaved layout).
    VH bf0, bf1;
#pragma unroll
    for (int v = 0; v < 8; ++v) {
      bf0.u[v] = Bsp[cur][(v + 8 * hh) * LDBW + wc * 32 + l16];
      bf1.u[v] = Bsp[cur][(v + 8 * hh) * LDBW + wc * 32 + 16 + l16];
    }
    acc0 = __builtin_amdgcn_wmma_f32_16x16x32_f16(false, af.v, false, bf0.v,
                                                  (short)0, acc0, false, false);
    acc1 = __builtin_amdgcn_wmma_f32_16x16x32_f16(false, af.v, false, bf1.v,
                                                  (short)0, acc1, false, false);

    asm volatile("s_wait_asynccnt 0x0" ::: "memory");
    __syncthreads();
    cur = nxt;
  }

  const float* a0 = (const float*)&acc0;
  const float* a1 = (const float*)&acc1;
#pragma unroll
  for (int v = 0; v < 8; ++v) {
    int row = tileO + wr * 16 + v + 8 * hh;
    int col = tileM + wc * 32 + l16;
    C[(size_t)row * M + col] = a0[v];
    C[(size_t)row * M + col + 16] = a1[v];
  }
}

// ------------------------- conv-BN(+relu) epilogues -------------------------
__global__ void chan_stats_k(const float* __restrict__ y, float* m, float* inv, int M) {
  __shared__ float rs[256], rq[256];
  int o = blockIdx.x;
  float s = 0.f, q = 0.f;
  for (int i = threadIdx.x; i < M; i += 256) {
    float v = y[(size_t)o * M + i];
    s += v; q += v * v;
  }
  rs[threadIdx.x] = s; rq[threadIdx.x] = q; __syncthreads();
  for (int st = 128; st > 0; st >>= 1) {
    if (threadIdx.x < st) { rs[threadIdx.x] += rs[threadIdx.x + st]; rq[threadIdx.x] += rq[threadIdx.x + st]; }
    __syncthreads();
  }
  if (threadIdx.x == 0) {
    float mm = rs[0] / (float)M;
    float var = rq[0] / (float)M - mm * mm;
    m[o] = mm; inv[o] = rsqrtf(var + BNEPS);
  }
}

__global__ void bn_relu_f16_k(const float* __restrict__ y, const float* m, const float* inv,
                              const float* g, const float* be, _Float16* __restrict__ Xn, int M) {
  int i = blockIdx.x * blockDim.x + threadIdx.x;
  int o = i / M, mc = i - o * M;
  float v = g[o] * (y[(size_t)o * M + mc] - m[o]) * inv[o] + be[o];
  Xn[(size_t)o * M + mc] = (_Float16)(v > 0.f ? v : 0.f);
}

__global__ void bn_relu_out_k(const float* __restrict__ y, const float* m, const float* inv,
                              const float* g, const float* be, float* __restrict__ out, int M) {
  int i = blockIdx.x * blockDim.x + threadIdx.x;
  int o = i / M, mc = i - o * M;
  int b = mc >> 11, n = mc & (NN - 1);
  float v = g[o] * (y[(size_t)o * M + mc] - m[o]) * inv[o] + be[o];
  out[((size_t)b * 420 + o) * NN + n] = v > 0.f ? v : 0.f;
}

// ============================== host launcher ===============================
extern "C" void kernel_launch(void* const* d_in, const int* in_sizes, int n_in,
                              void* d_out, int out_size, void* d_ws, size_t ws_size,
                              hipStream_t stream) {
  (void)in_sizes; (void)n_in; (void)out_size; (void)ws_size;
  const float* coord = (const float*)d_in[0];   // (B,1,3,N) == (B,3,N)
  const float* nrm   = (const float*)d_in[1];
  const int*   cat   = (const int*)d_in[2];
  const float* wf0 = (const float*)d_in[3];  const float* wd0 = (const float*)d_in[4];
  const float* g0  = (const float*)d_in[5];  const float* b0  = (const float*)d_in[6];
  const float* wf1 = (const float*)d_in[7];  const float* wd1 = (const float*)d_in[8];
  const float* g1  = (const float*)d_in[9];  const float* b1  = (const float*)d_in[10];
  const float* wp1 = (const float*)d_in[11];
  const float* wf2 = (const float*)d_in[12]; const float* wd2 = (const float*)d_in[13];
  const float* g2  = (const float*)d_in[14]; const float* b2  = (const float*)d_in[15];
  const float* wf3 = (const float*)d_in[16]; const float* wd3 = (const float*)d_in[17];
  const float* g3  = (const float*)d_in[18]; const float* b3  = (const float*)d_in[19];
  const float* wp2 = (const float*)d_in[20];
  const float* wf4 = (const float*)d_in[21]; const float* wd4 = (const float*)d_in[22];
  const float* g4  = (const float*)d_in[23]; const float* b4  = (const float*)d_in[24];
  const float* wv1f = (const float*)d_in[25]; const float* wv1d = (const float*)d_in[26];
  const float* wv2f = (const float*)d_in[27]; const float* wv2d = (const float*)d_in[28];
  const float* w3   = (const float*)d_in[29];
  const float* ws1 = (const float*)d_in[30];  /* cb1 (d_in[31]) cancels in BN */
  const float* sg1 = (const float*)d_in[32]; const float* sb1 = (const float*)d_in[33];
  const float* ws2 = (const float*)d_in[34];
  const float* sg2 = (const float*)d_in[36]; const float* sb2 = (const float*)d_in[37];
  const float* ws3 = (const float*)d_in[38];
  const float* sg3 = (const float*)d_in[40]; const float* sb3 = (const float*)d_in[41];

  float* out   = (float*)d_out;
  float* eqv   = out;                  // (B,420,3,N) = 10,321,920
  float* mf    = out + 10321920;       // (B,420,3)   = 5,040
  float* invo  = out + 10326960;       // (B,420,N)   = 3,440,640
  float* invgl = out + 13767600;       // (B,420,2)   = 3,360

  char* base = (char*)d_ws; size_t off = 0;
  auto alloc = [&](size_t bytes) -> char* {
    off = (off + 255) & ~(size_t)255;
    char* p = base + off; off += bytes; return p;
  };
  const int M = BB * NN;  // 8192 columns for the invariant head

  int* idx20  = (int*)alloc((size_t)BB * NN * 20 * 4);
  int* idx4a  = (int*)alloc((size_t)BB * 512 * 4 * 4);
  int* idx4b  = (int*)alloc((size_t)BB * 128 * 4 * 4);
  int* i1     = (int*)alloc((size_t)BB * NN * 4);
  int* i2     = (int*)alloc((size_t)BB * NN * 4);
  float* ssum = (float*)alloc(2048 * 4);
  float* ssq  = (float*)alloc(2048 * 4);
  float* smn  = (float*)alloc(2048 * 4);
  float* siv  = (float*)alloc(2048 * 4);
  float* x0b  = (float*)alloc((size_t)BB * 42 * 3 * NN * 4);
  float* x1b  = (float*)alloc((size_t)BB * 42 * 3 * NN * 4);
  float* pbuf = (float*)alloc((size_t)BB * 42 * 3 * NN * 4);   // largest stage
  float* dbuf = (float*)alloc((size_t)BB * 42 * 3 * NN * 4);
  float* x2p  = (float*)alloc((size_t)BB * 42 * 3 * 512 * 4);
  float* x2b  = (float*)alloc((size_t)BB * 84 * 3 * 512 * 4);
  float* x3b  = (float*)alloc((size_t)BB * 84 * 3 * 512 * 4);
  float* x4p  = (float*)alloc((size_t)BB * 84 * 3 * 128 * 4);
  float* x4b  = (float*)alloc((size_t)BB * 168 * 3 * 128 * 4);
  float* z1   = (float*)alloc((size_t)BB * 210 * 3 * 4);
  float* z2   = (float*)alloc((size_t)BB * 84 * 3 * 4);
  _Float16* Xa  = (_Float16*)alloc((size_t)1280 * M * 2);
  _Float16* Xb  = (_Float16*)alloc((size_t)1024 * M * 2);
  _Float16* Xc  = (_Float16*)alloc((size_t)512 * M * 2);
  _Float16* W1h = (_Float16*)alloc((size_t)1024 * 1280 * 2);
  _Float16* W2h = (_Float16*)alloc((size_t)512 * 1024 * 2);
  _Float16* W3h = (_Float16*)alloc((size_t)448 * 512 * 2);
  float* y1 = (float*)alloc((size_t)1024 * M * 4);
  float* y2 = (float*)alloc((size_t)512 * M * 4);
  float* y3 = (float*)alloc((size_t)448 * M * 4);

  auto gz = [](int n) { return (n + 255) / 256; };

  // ---- VN block helper (two pass: lin+stats, then bn+leaky) ----
  auto vn_stage = [&](const float* xin, const float* wf, const float* wd,
                      const float* g, const float* bb, float* xout,
                      int Cin, int Cout, int Np) {
    zero_f32_k<<<gz(Cout), 256, 0, stream>>>(ssum, Cout);
    zero_f32_k<<<gz(Cout), 256, 0, stream>>>(ssq, Cout);
    vn_lin_pair_k<<<dim3(Cout, BB), 256, 0, stream>>>(xin, wf, wd, pbuf, dbuf,
                                                      ssum, ssq, Cin, Cout, Np);
    stats_fin_k<<<gz(Cout), 256, 0, stream>>>(ssum, ssq, smn, siv, Cout,
                                              (float)(BB * Np));
    vn_bn_leaky_k<<<BB * Cout, 256, 0, stream>>>(pbuf, dbuf, g, bb, smn, siv,
                                                 xout, Cout, Np);
  };

  // ---- stage 0: surface feature + VN block (k=20) ----
  knn_kernel<20><<<gz(BB * NN), 256, 0, stream>>>(coord, NN, 1, 1, NN, idx20);
  zero_f32_k<<<gz(42), 256, 0, stream>>>(ssum, 42);
  zero_f32_k<<<gz(42), 256, 0, stream>>>(ssq, 42);
  surf_stats_k<<<dim3(42, BB), 256, 0, stream>>>(coord, idx20, wf0, ssum, ssq);
  stats_fin_k<<<gz(42), 256, 0, stream>>>(ssum, ssq, smn, siv, 42,
                                          (float)(BB * NN * 20));
  surf_apply_k<<<dim3(42, BB), 256, 0, stream>>>(coord, idx20, wf0, wd0, g0, b0,
                                                 smn, siv, x0b);

  // ---- x1 ----
  vn_stage(x0b, wf1, wd1, g1, b1, x1b, 42, 42, NN);

  // ---- pool 1 (2048 -> 512) + x2, x3 ----
  knn_kernel<4><<<gz(BB * 512), 256, 0, stream>>>(coord, NN, 1, 4, 512, idx4a);
  pool_down_k<<<BB * 512, 128, 0, stream>>>(x1b, idx4a, wp1, x2p, 42, NN, 512);
  vn_stage(x2p, wf2, wd2, g2, b2, x2b, 42, 84, 512);
  vn_stage(x2b, wf3, wd3, g3, b3, x3b, 84, 84, 512);

  // ---- pool 2 (512 -> 128) + x4 ----
  knn_kernel<4><<<gz(BB * 128), 256, 0, stream>>>(coord, 512, 4, 16, 128, idx4b);
  pool_down_k<<<BB * 128, 128, 0, stream>>>(x3b, idx4b, wp2, x4p, 84, 512, 128);
  vn_stage(x4p, wf4, wd4, g4, b4, x4b, 84, 168, 128);

  // ---- upsample indices + eqv + mean_feat ----
  nearest_kernel<<<gz(BB * NN), 256, 0, stream>>>(coord, 512, 4, i1);
  nearest_kernel<<<gz(BB * NN), 256, 0, stream>>>(coord, 128, 16, i2);
  eqv_assemble_k<<<gz(BB * 420 * NN), 256, 0, stream>>>(x0b, x1b, x2b, x3b, x4b,
                                                        i1, i2, eqv);
  mean_feat_k<<<BB * 420 * 3, 128, 0, stream>>>(eqv, mf);

  // ---- global z pipeline + inv_gl ----
  vn_block_small_k<<<gz(BB * 210), 256, 0, stream>>>(mf, wv1f, wv1d, z1, 420, 210);
  vn_block_small_k<<<gz(BB * 84), 256, 0, stream>>>(z1, wv2f, wv2d, z2, 210, 84);
  inv_gl_k<<<gz(BB * 420), 256, 0, stream>>>(mf, z2, w3, invgl);

  // ---- invariant head: 3 WMMA GEMMs (f16 in, f32 accum) + BN/ReLU ----
  zero_f16_k<<<gz(1280 * M), 256, 0, stream>>>(Xa, 1280 * M);  // zero K-pad rows
  build_inv_input_k<<<gz(1267 * M), 256, 0, stream>>>(nrm, cat, mf, invgl, coord, Xa);
  pad_w_k<<<gz(1024 * 1280), 256, 0, stream>>>(ws1, W1h, 1024, 1267, 1024, 1280);
  pad_w_k<<<gz(512 * 1024), 256, 0, stream>>>(ws2, W2h, 512, 1024, 512, 1024);
  pad_w_k<<<gz(448 * 512), 256, 0, stream>>>(ws3, W3h, 420, 512, 448, 512);

  gemm_f16_wmma_k<<<dim3(M / 64, 1024 / 64), 256, 0, stream>>>(W1h, Xa, y1, 1280, M);
  chan_stats_k<<<1024, 256, 0, stream>>>(y1, smn, siv, M);
  bn_relu_f16_k<<<(1024 * M) / 256, 256, 0, stream>>>(y1, smn, siv, sg1, sb1, Xb, M);

  gemm_f16_wmma_k<<<dim3(M / 64, 512 / 64), 256, 0, stream>>>(W2h, Xb, y2, 1024, M);
  chan_stats_k<<<512, 256, 0, stream>>>(y2, smn, siv, M);
  bn_relu_f16_k<<<(512 * M) / 256, 256, 0, stream>>>(y2, smn, siv, sg2, sb2, Xc, M);

  gemm_f16_wmma_k<<<dim3(M / 64, 448 / 64), 256, 0, stream>>>(W3h, Xc, y3, 512, M);
  chan_stats_k<<<420, 256, 0, stream>>>(y3, smn, siv, M);
  bn_relu_out_k<<<(420 * M) / 256, 256, 0, stream>>>(y3, smn, siv, sg3, sb3, invo, M);
}
